// MatchingNetwork_20538533609521
// MI455X (gfx1250) — matrix operand order repaired and checked by reading, compiled, not run on hardware
//
#include <hip/hip_runtime.h>
#include <hip/hip_bf16.h>
#include <math.h>

typedef __attribute__((ext_vector_type(16))) _Float16 v16h;
typedef __attribute__((ext_vector_type(8)))  _Float16 v8h;
typedef __attribute__((ext_vector_type(8)))  float    v8f;

#define N_SUP 25
#define BATCH 128
#define NIMG  3328   // 25*128 support + 128 target
#define NGRP  26

__device__ __forceinline__ float sigmoidf(float x) { return 1.f / (1.f + __expf(-x)); }

// ---------------------------------------------------------------------------
// WMMA fragment loaders (CDNA5 16-bit layouts, wave32)
// A: 16x32 (MxK), row-major [row][Kpad].
//   lane<16: row=lane,    k = {0..7, 16..23}
//   lane>=16: row=lane-16, k = {8..15, 24..31}
// ---------------------------------------------------------------------------
__device__ __forceinline__ v16h load_a_frag(const _Float16* __restrict__ A,
                                            int Kpad, int row, int ks, int lane) {
  int kb = ks * 32 + ((lane >> 4) << 3);      // 0 or 8
  v8h lo = *(const v8h*)(A + (size_t)row * Kpad + kb);
  v8h hi = *(const v8h*)(A + (size_t)row * Kpad + kb + 16);
  v16h a;
#pragma unroll
  for (int i = 0; i < 8; ++i) { a[i] = lo[i]; a[8 + i] = hi[i]; }
  return a;
}

// B: 32x16 (KxN) stored in LDS as [n][Kpad]:
//   lane<16: col=lane, K={0..7,16..23}; lane>=16: col=lane-16, K={8..15,24..31}.
__device__ __forceinline__ v16h load_b_frag(const _Float16* Bs,
                                            int Kpad, int col, int ks, int lane) {
  int kb = ks * 32 + ((lane >> 4) << 3);      // 0 or 8
  v8h lo = *(const v8h*)(Bs + (size_t)col * Kpad + kb);
  v8h hi = *(const v8h*)(Bs + (size_t)col * Kpad + kb + 16);
  v16h b;
#pragma unroll
  for (int i = 0; i < 8; ++i) { b[i] = lo[i]; b[8 + i] = hi[i]; }
  return b;
}

// Async global->LDS copy of one 16B chunk (CDNA5 TDM-class data path).
__device__ __forceinline__ void async_copy_b128(const void* gsrc, void* ldsdst) {
  unsigned ldsoff = (unsigned)(unsigned long long)ldsdst;   // low 32b = LDS offset
  unsigned long long gaddr = (unsigned long long)gsrc;
  asm volatile("global_load_async_to_lds_b128 %0, %1, off"
               :: "v"(ldsoff), "v"(gaddr) : "memory");
}
__device__ __forceinline__ void wait_async() {
  asm volatile("s_wait_asynccnt 0" ::: "memory");
}

// ---------------------------------------------------------------------------
// Conv3x3 (pad=1) as WMMA GEMM: D[64, HW] = W[64, CIN*9] x im2col[CIN*9, HW]
// One workgroup per image (4 wave32). Whole input image async-staged to LDS
// once; loop over 32-position panels; 2 N-tiles per wave per panel so each A
// fragment feeds 2 WMMAs. Fused bias+ReLU, f16 store of pre-BN activation,
// per-image BN sum/sumsq accumulated in LDS, flushed with 128 global atomics.
// ---------------------------------------------------------------------------
template <int CIN, int H, int W, bool FIRST>
__global__ __launch_bounds__(128) void conv_wmma(
    const _Float16* __restrict__ A,     // [64][KPAD] f16 weights
    const float* __restrict__ bias,     // [64]
    const float* __restrict__ s,        // FIRST: [128][25][1][28][28]
    const float* __restrict__ tgt,      // FIRST: [128][1][28][28]
    const _Float16* __restrict__ xin,   // !FIRST: [NIMG][CIN][H][W]
    _Float16* __restrict__ y,           // [NIMG][64][H][W]
    float* __restrict__ stats)          // [NGRP][64][2] sum/sumsq
{
  constexpr int HW = H * W;
  constexpr int K = CIN * 9;
  constexpr int KPAD = (K + 31) / 32 * 32;
  constexpr int NPANEL = (HW + 31) / 32;              // 32 positions / panel
  constexpr int XBYTES = FIRST ? HW * 4 : CIN * HW * 2;
  constexpr int XCHUNK = XBYTES / 16;                 // all cases 16B-multiples
  __shared__ __align__(64) _Float16 Bs[32 * KPAD];
  __shared__ __align__(64) unsigned char Xs[XBYTES];
  __shared__ float ssum[64], ssq[64];

  const int img = blockIdx.x;
  const int tid = threadIdx.x;
  if (tid < 64) { ssum[tid] = 0.f; ssq[tid] = 0.f; }

  // --- Stage the whole input image into LDS via async b128 copies ---
  const unsigned char* src;
  if (FIRST) {
    if (img < N_SUP * BATCH) {
      int nn = img >> 7, b = img & 127;               // img = n*128 + b
      src = (const unsigned char*)(s + (size_t)(b * N_SUP + nn) * HW);
    } else {
      src = (const unsigned char*)(tgt + (size_t)(img - N_SUP * BATCH) * HW);
    }
  } else {
    src = (const unsigned char*)(xin + (size_t)img * CIN * HW);
  }
  for (int c = tid; c < XCHUNK; c += 128)
    async_copy_b128(src + (size_t)c * 16, Xs + c * 16);
  wait_async();
  __syncthreads();

  const float*    Xf = (const float*)Xs;      // FIRST staging is f32
  const _Float16* Xh = (const _Float16*)Xs;   // later blocks are f16
  const int lane = tid & 31;
  const int mtile = tid >> 5;
  const int row = mtile * 16 + (lane & 15);
  const int col = lane & 15;
  const int lh = lane >> 4;

  for (int pt = 0; pt < NPANEL; ++pt) {
    // Build im2col panel B[32 positions][KPAD] (LDS -> LDS)
    for (int idx = tid; idx < 32 * KPAD; idx += 128) {
      int n = idx / KPAD, k = idx - n * KPAD;
      int pos = pt * 32 + n;
      float v = 0.f;
      if (k < K && pos < HW) {
        int ci = k / 9, r = k - ci * 9;
        int oy = pos / W, ox = pos - oy * W;
        int iy = oy + r / 3 - 1, ix = ox + (r % 3) - 1;
        if (iy >= 0 && iy < H && ix >= 0 && ix < W)
          v = FIRST ? Xf[iy * W + ix] : (float)Xh[ci * HW + iy * W + ix];
      }
      Bs[idx] = (_Float16)v;
    }
    __syncthreads();

    v8f z = {0.f, 0.f, 0.f, 0.f, 0.f, 0.f, 0.f, 0.f};
    v8f acc0 = z, acc1 = z;
#pragma unroll 2
    for (int ks = 0; ks < KPAD / 32; ++ks) {
      v16h a  = load_a_frag(A, KPAD, row, ks, lane);
      v16h b0 = load_b_frag(Bs, KPAD, col, ks, lane);
      v16h b1 = load_b_frag(Bs, KPAD, col + 16, ks, lane);
      acc0 = __builtin_amdgcn_wmma_f32_16x16x32_f16(false, a, false, b0,
                                                    (short)0, acc0, false, false);
      acc1 = __builtin_amdgcn_wmma_f32_16x16x32_f16(false, a, false, b1,
                                                    (short)0, acc1, false, false);
    }

    // Epilogue: D layout — VGPR v, lane: M = v + 8*(lane>>4), N = lane&15
#pragma unroll
    for (int v = 0; v < 8; ++v) {
      int ch = mtile * 16 + v + 8 * lh;
      float bval = bias[ch];
#pragma unroll
      for (int t = 0; t < 2; ++t) {
        int pos = pt * 32 + t * 16 + col;
        if (pos < HW) {
          float val = fmaxf((t ? acc1[v] : acc0[v]) + bval, 0.f);
          y[((size_t)img * 64 + ch) * HW + pos] = (_Float16)val;
          atomicAdd(&ssum[ch], val);        // LDS f32 atomics (ds_add_f32)
          atomicAdd(&ssq[ch], val * val);
        }
      }
    }
    __syncthreads();
  }

  if (tid < 64) {
    int g = img >> 7;                       // 26 groups of 128 images
    atomicAdd(&stats[(g * 64 + tid) * 2 + 0], ssum[tid]);
    atomicAdd(&stats[(g * 64 + tid) * 2 + 1], ssq[tid]);
  }
}

// ---------------------------------------------------------------------------
// BN finalize: per (group, channel) scale/shift from accumulated stats.
// ---------------------------------------------------------------------------
__global__ void bn_finalize(const float* __restrict__ stats,
                            const float* __restrict__ gamma,
                            const float* __restrict__ beta, float count,
                            float* __restrict__ scale, float* __restrict__ shift) {
  int i = blockIdx.x * blockDim.x + threadIdx.x;
  if (i >= NGRP * 64) return;
  int ch = i & 63;
  float mean = stats[i * 2] / count;
  float var = stats[i * 2 + 1] / count - mean * mean;
  float sc = gamma[ch] * rsqrtf(var + 1e-5f);
  scale[i] = sc;
  shift[i] = beta[ch] - mean * sc;
}

// ---------------------------------------------------------------------------
// BN affine + 2x2 maxpool (stride 2, VALID). Affine before max (ref order).
// ---------------------------------------------------------------------------
template <int H, int W, typename OutT>
__global__ __launch_bounds__(256) void bn_pool(const _Float16* __restrict__ y,
                                               const float* __restrict__ scale,
                                               const float* __restrict__ shift,
                                               OutT* __restrict__ out) {
  constexpr int Ho = H / 2, Wo = W / 2;
  const size_t total = (size_t)NIMG * 64 * Ho * Wo;
  for (size_t i = blockIdx.x * (size_t)blockDim.x + threadIdx.x; i < total;
       i += (size_t)gridDim.x * blockDim.x) {
    int ow = (int)(i % Wo);
    size_t r = i / Wo;
    int oh = (int)(r % Ho); r /= Ho;
    int ch = (int)(r % 64);
    int img = (int)(r / 64);
    int g = img >> 7;
    float sc = scale[g * 64 + ch], sh = shift[g * 64 + ch];
    const _Float16* base = y + ((size_t)img * 64 + ch) * H * W;
    float m = -3.4e38f;
#pragma unroll
    for (int dy = 0; dy < 2; ++dy)
#pragma unroll
      for (int dx = 0; dx < 2; ++dx)
        m = fmaxf(m, (float)base[(2 * oh + dy) * W + (2 * ow + dx)] * sc + sh);
    out[i] = (OutT)m;
  }
}

// ---------------------------------------------------------------------------
// fce_g: bidirectional LSTM (hidden 32) over N=25 support embeddings + residual.
// One workgroup per direction; h in LDS, c in registers.
// ---------------------------------------------------------------------------
__global__ __launch_bounds__(256) void fce_g_kernel(
    const float* __restrict__ E,   // [NIMG][64], support at img = n*128+b
    const float* __restrict__ wih_f, const float* __restrict__ whh_f,
    const float* __restrict__ bih_f, const float* __restrict__ bhh_f,
    const float* __restrict__ wih_b, const float* __restrict__ whh_b,
    const float* __restrict__ bih_b, const float* __restrict__ bhh_b,
    float* __restrict__ sup)       // [25][128][64] = E + concat(hf, hb)
{
  __shared__ float sh[BATCH * 32];
  __shared__ float shn[BATCH * 32];
  const int tid = threadIdx.x;
  const int dir = blockIdx.x;
  const float* wih = dir ? wih_b : wih_f;
  const float* whh = dir ? whh_b : whh_f;
  const float* bih = dir ? bih_b : bih_f;
  const float* bhh = dir ? bhh_b : bhh_f;

  float creg[16];
#pragma unroll
  for (int p = 0; p < 16; ++p) creg[p] = 0.f;
  for (int idx = tid; idx < BATCH * 32; idx += 256) sh[idx] = 0.f;
  __syncthreads();

  for (int st = 0; st < N_SUP; ++st) {
    int t = dir ? (N_SUP - 1 - st) : st;
#pragma unroll 1
    for (int p = 0; p < 16; ++p) {
      int it = p * 256 + tid;
      int b = it >> 5, j = it & 31;
      const float* xb = E + (size_t)(t * BATCH + b) * 64;
      const float* hb = sh + b * 32;
      float z[4];
#pragma unroll
      for (int g = 0; g < 4; ++g) {
        int gate = g * 32 + j;                 // torch gate order i,f,g,o
        float a = bih[gate] + bhh[gate];
        const float* wr = wih + gate * 64;
        for (int d = 0; d < 64; ++d) a += xb[d] * wr[d];
        const float* ur = whh + gate * 32;
        for (int d = 0; d < 32; ++d) a += hb[d] * ur[d];
        z[g] = a;
      }
      float cn = sigmoidf(z[1]) * creg[p] + sigmoidf(z[0]) * tanhf(z[2]);
      float hn = sigmoidf(z[3]) * tanhf(cn);
      creg[p] = cn;
      shn[it] = hn;
      int dcol = dir ? (32 + j) : j;
      size_t o = (size_t)(t * BATCH + b) * 64 + dcol;
      sup[o] = E[o] + hn;                      // residual
    }
    __syncthreads();
    for (int idx = tid; idx < BATCH * 32; idx += 256) sh[idx] = shn[idx];
    __syncthreads();
  }
}

// ---------------------------------------------------------------------------
// fce_f (10-step attention LSTM over support memory) + cosine classifier.
// Single workgroup; h and h_k in LDS (64KB), c in registers.
// ---------------------------------------------------------------------------
__global__ __launch_bounds__(256) void fce_f_kernel(
    const float* __restrict__ E,    // sample at E + 3200*64
    const float* __restrict__ sup,  // memory [25][128][64]
    const float* __restrict__ wih, const float* __restrict__ whh,
    const float* __restrict__ bih, const float* __restrict__ bhh,
    float* __restrict__ out)        // [128][25]
{
  __shared__ float sh[BATCH * 64];
  __shared__ float shk[BATCH * 64];
  const int tid = threadIdx.x;
  const float* X = E + (size_t)N_SUP * BATCH * 64;

  float creg[32];
#pragma unroll
  for (int p = 0; p < 32; ++p) creg[p] = 0.f;
  for (int idx = tid; idx < BATCH * 64; idx += 256) sh[idx] = 0.f;
  __syncthreads();

  for (int step = 0; step < 10; ++step) {
    // lstm_step(x, h, c) -> h_cap, c ; h_k = h_cap + x
#pragma unroll 1
    for (int p = 0; p < 32; ++p) {
      int it = p * 256 + tid;
      int b = it >> 6, j = it & 63;
      const float* xb = X + b * 64;
      const float* hb = sh + b * 64;
      float z[4];
#pragma unroll
      for (int g = 0; g < 4; ++g) {
        int gate = g * 64 + j;
        float a = bih[gate] + bhh[gate];
        const float* wr = wih + gate * 64;
        const float* ur = whh + gate * 64;
        for (int d = 0; d < 64; ++d) a += xb[d] * wr[d] + hb[d] * ur[d];
        z[g] = a;
      }
      float cn = sigmoidf(z[1]) * creg[p] + sigmoidf(z[0]) * tanhf(z[2]);
      float hcap = sigmoidf(z[3]) * tanhf(cn);
      creg[p] = cn;
      shk[it] = hcap + xb[j];
    }
    __syncthreads();
    // attention read with OLD h: softmax over support dim of h*mem, r = sum(attn*mem)
#pragma unroll 1
    for (int p = 0; p < 32; ++p) {
      int it = p * 256 + tid;
      int b = it >> 6, d = it & 63;
      float hv = sh[it];
      const float* mp = sup + (size_t)b * 64 + d;   // stride 128*64 per n
      float mmax = -3.4e38f;
      for (int n = 0; n < N_SUP; ++n)
        mmax = fmaxf(mmax, hv * mp[(size_t)n * (BATCH * 64)]);
      float ssum = 0.f, rnum = 0.f;
      for (int n = 0; n < N_SUP; ++n) {
        float mv = mp[(size_t)n * (BATCH * 64)];
        float e = __expf(hv * mv - mmax);
        ssum += e;
        rnum += e * mv;
      }
      sh[it] = shk[it] + rnum / ssum;               // h = h_k + r_k
    }
    __syncthreads();
  }

  // cosine classifier: out[b][n] = <sup[n,b], hk[b]> / (|sup||hk|)
  for (int i = tid; i < BATCH * N_SUP; i += 256) {
    int b = i / N_SUP, n = i - b * N_SUP;
    const float* tv = shk + b * 64;
    const float* sv = sup + ((size_t)n * BATCH + b) * 64;
    float dot = 0.f, nt = 0.f, nx = 0.f;
    for (int d = 0; d < 64; ++d) {
      float a = tv[d], c = sv[d];
      dot += a * c; nt += a * a; nx += c * c;
    }
    out[b * N_SUP + n] =
        dot * rsqrtf(fmaxf(nt, 1e-10f)) * rsqrtf(fmaxf(nx, 1e-10f));
  }
}

// ---------------------------------------------------------------------------
// Utility kernels
// ---------------------------------------------------------------------------
__global__ void convert_pad(const float* __restrict__ src, _Float16* __restrict__ dst,
                            int rows, int ksrc, int kdst) {
  int total = rows * kdst;
  for (int i = blockIdx.x * blockDim.x + threadIdx.x; i < total;
       i += gridDim.x * blockDim.x) {
    int r = i / kdst, k = i - r * kdst;
    dst[i] = (_Float16)(k < ksrc ? src[r * ksrc + k] : 0.f);
  }
}

__global__ void zero_f32(float* __restrict__ p, int n) {
  for (int i = blockIdx.x * blockDim.x + threadIdx.x; i < n;
       i += gridDim.x * blockDim.x)
    p[i] = 0.f;
}

// ---------------------------------------------------------------------------
extern "C" void kernel_launch(void* const* d_in, const int* in_sizes, int n_in,
                              void* d_out, int out_size, void* d_ws, size_t ws_size,
                              hipStream_t stream) {
  (void)in_sizes; (void)n_in; (void)out_size; (void)ws_size;
  const float* s      = (const float*)d_in[0];
  const float* target = (const float*)d_in[2];
  const float* cw[4]  = {(const float*)d_in[4],  (const float*)d_in[8],
                         (const float*)d_in[12], (const float*)d_in[16]};
  const float* cb[4]  = {(const float*)d_in[5],  (const float*)d_in[9],
                         (const float*)d_in[13], (const float*)d_in[17]};
  const float* bng[4] = {(const float*)d_in[6],  (const float*)d_in[10],
                         (const float*)d_in[14], (const float*)d_in[18]};
  const float* bnb[4] = {(const float*)d_in[7],  (const float*)d_in[11],
                         (const float*)d_in[15], (const float*)d_in[19]};
  const float* g_wih_f = (const float*)d_in[20];
  const float* g_whh_f = (const float*)d_in[21];
  const float* g_bih_f = (const float*)d_in[22];
  const float* g_bhh_f = (const float*)d_in[23];
  const float* g_wih_b = (const float*)d_in[24];
  const float* g_whh_b = (const float*)d_in[25];
  const float* g_bih_b = (const float*)d_in[26];
  const float* g_bhh_b = (const float*)d_in[27];
  const float* f_wih = (const float*)d_in[28];
  const float* f_whh = (const float*)d_in[29];
  const float* f_bih = (const float*)d_in[30];
  const float* f_bhh = (const float*)d_in[31];
  float* out = (float*)d_out;

  // Workspace bump allocator (256B aligned)
  char* base = (char*)d_ws;
  size_t off = 0;
  auto alloc = [&](size_t bytes) -> void* {
    off = (off + 255) & ~(size_t)255;
    void* p = base + off;
    off += bytes;
    return p;
  };
  _Float16* wA1  = (_Float16*)alloc((size_t)64 * 32 * 2);
  _Float16* wA2  = (_Float16*)alloc((size_t)64 * 576 * 2);
  _Float16* wA3  = (_Float16*)alloc((size_t)64 * 576 * 2);
  _Float16* wA4  = (_Float16*)alloc((size_t)64 * 576 * 2);
  _Float16* ybuf = (_Float16*)alloc((size_t)NIMG * 64 * 28 * 28 * 2); // pre-pool (reused)
  _Float16* pbuf = (_Float16*)alloc((size_t)NIMG * 64 * 14 * 14 * 2); // pooled  (reused)
  float* E     = (float*)alloc((size_t)NIMG * 64 * 4);
  float* supB  = (float*)alloc((size_t)N_SUP * BATCH * 64 * 4);
  float* stats = (float*)alloc((size_t)4 * NGRP * 64 * 2 * 4);
  float* bnsc  = (float*)alloc((size_t)4 * NGRP * 64 * 4);
  float* bnsh  = (float*)alloc((size_t)4 * NGRP * 64 * 4);
  const int SEG = NGRP * 64;     // per-block stride into scale/shift
  const int SST = NGRP * 64 * 2; // per-block stride into stats

  zero_f32<<<(4 * SST + 255) / 256, 256, 0, stream>>>(stats, 4 * SST);
  convert_pad<<<16, 256, 0, stream>>>(cw[0], wA1, 64, 9, 32);
  convert_pad<<<144, 256, 0, stream>>>(cw[1], wA2, 64, 576, 576);
  convert_pad<<<144, 256, 0, stream>>>(cw[2], wA3, 64, 576, 576);
  convert_pad<<<144, 256, 0, stream>>>(cw[3], wA4, 64, 576, 576);

  // Block 1: 1x28x28 -> 64x28x28 -> pool 14x14
  conv_wmma<1, 28, 28, true><<<NIMG, 128, 0, stream>>>(
      wA1, cb[0], s, target, nullptr, ybuf, stats + 0 * SST);
  bn_finalize<<<(SEG + 255) / 256, 256, 0, stream>>>(
      stats + 0 * SST, bng[0], bnb[0], 128.f * 784.f, bnsc + 0 * SEG, bnsh + 0 * SEG);
  bn_pool<28, 28, _Float16><<<4096, 256, 0, stream>>>(ybuf, bnsc + 0 * SEG,
                                                      bnsh + 0 * SEG, pbuf);
  // Block 2: 64x14x14 -> pool 7x7
  conv_wmma<64, 14, 14, false><<<NIMG, 128, 0, stream>>>(
      wA2, cb[1], nullptr, nullptr, pbuf, ybuf, stats + 1 * SST);
  bn_finalize<<<(SEG + 255) / 256, 256, 0, stream>>>(
      stats + 1 * SST, bng[1], bnb[1], 128.f * 196.f, bnsc + 1 * SEG, bnsh + 1 * SEG);
  bn_pool<14, 14, _Float16><<<2048, 256, 0, stream>>>(ybuf, bnsc + 1 * SEG,
                                                      bnsh + 1 * SEG, pbuf);
  // Block 3: 64x7x7 -> pool 3x3
  conv_wmma<64, 7, 7, false><<<NIMG, 128, 0, stream>>>(
      wA3, cb[2], nullptr, nullptr, pbuf, ybuf, stats + 2 * SST);
  bn_finalize<<<(SEG + 255) / 256, 256, 0, stream>>>(
      stats + 2 * SST, bng[2], bnb[2], 128.f * 49.f, bnsc + 2 * SEG, bnsh + 2 * SEG);
  bn_pool<7, 7, _Float16><<<1024, 256, 0, stream>>>(ybuf, bnsc + 2 * SEG,
                                                    bnsh + 2 * SEG, pbuf);
  // Block 4: 64x3x3 -> pool 1x1 -> embeddings (f32)
  conv_wmma<64, 3, 3, false><<<NIMG, 128, 0, stream>>>(
      wA4, cb[3], nullptr, nullptr, pbuf, ybuf, stats + 3 * SST);
  bn_finalize<<<(SEG + 255) / 256, 256, 0, stream>>>(
      stats + 3 * SST, bng[3], bnb[3], 128.f * 9.f, bnsc + 3 * SEG, bnsh + 3 * SEG);
  bn_pool<3, 3, float><<<832, 256, 0, stream>>>(ybuf, bnsc + 3 * SEG,
                                                bnsh + 3 * SEG, E);

  // FCE g (bi-LSTM + residual), FCE f (attention LSTM) + cosine head
  fce_g_kernel<<<2, 256, 0, stream>>>(E, g_wih_f, g_whh_f, g_bih_f, g_bhh_f,
                                      g_wih_b, g_whh_b, g_bih_b, g_bhh_b, supB);
  fce_f_kernel<<<1, 256, 0, stream>>>(E, supB, f_wih, f_whh, f_bih, f_bhh, out);
}
